// MoE_90280212562392
// MI455X (gfx1250) — matrix operand order repaired and checked by reading, not compiled, not run on hardware
//
#include <hip/hip_runtime.h>
#include <hip/hip_bf16.h>
#include <math.h>

// ---------------- problem constants (from reference) ----------------
#define NE      8       // experts
#define KTOP    2       // top-k
#define DIM     1024    // embedding
#define NTOK    8192    // B*S tokens
#define TILE_M  128     // tokens per expert tile
#define NTILES  64      // NTOK / TILE_M (max tiles per expert)
#define NCH     8       // n-chunks of 128 over DIM
#define HSTRIDE (DIM + 8)   // padded LDS h row stride (bf16 elems)
#define BSTRIDE 40          // padded transposed-B panel stride (32 K + pad), keeps 16B align

#define AS3 __attribute__((address_space(3)))

typedef __attribute__((ext_vector_type(16))) __bf16 v16bf;
typedef __attribute__((ext_vector_type(8)))  __bf16 bf16x8;
typedef __attribute__((ext_vector_type(4)))  __bf16 bf16x4;
typedef __attribute__((ext_vector_type(8)))  float  v8f;
typedef __attribute__((ext_vector_type(4)))  int    v4i;

typedef AS3 __bf16  lbf16;     // LDS bf16
typedef AS3 bf16x8  l_bf16x8;  // LDS 16-byte vector

// ---------------- gfx1250 async global->LDS (guarded) ----------------
#if defined(__has_builtin)
#if __has_builtin(__builtin_amdgcn_global_load_async_to_lds_b128)
#define HAVE_ASYNC_LDS 1
#endif
#endif
#ifndef HAVE_ASYNC_LDS
#define HAVE_ASYNC_LDS 0
#endif

#if HAVE_ASYNC_LDS
typedef __attribute__((address_space(1))) v4i gv4i;   // global
typedef AS3 v4i lv4i;                                 // LDS

__device__ __forceinline__ void async_copy_b128(const __bf16* g, lbf16* l) {
    __builtin_amdgcn_global_load_async_to_lds_b128((gv4i*)g, (lv4i*)l, 0, 0);
}
__device__ __forceinline__ void wait_async_le2() {
#if __has_builtin(__builtin_amdgcn_s_wait_asynccnt)
    __builtin_amdgcn_s_wait_asynccnt(2);
#else
    asm volatile("s_wait_asynccnt 0x2" ::: "memory");
#endif
}
__device__ __forceinline__ void wait_async_le0() {
#if __has_builtin(__builtin_amdgcn_s_wait_asynccnt)
    __builtin_amdgcn_s_wait_asynccnt(0);
#else
    asm volatile("s_wait_asynccnt 0x0" ::: "memory");
#endif
}
#endif

// ---------------- fragment helpers (layouts per CDNA5 ISA 7.12.2) ----------------

// A 16x32 bf16 fragment from a global bf16 row.
__device__ __forceinline__ v16bf load_a_g(const __bf16* row, int k0, int hl) {
    const __bf16* p = row + k0 + hl * 8;
    bf16x8 lo = *(const bf16x8*)p;
    bf16x8 hi = *(const bf16x8*)(p + 16);
    v16bf a;
#pragma unroll
    for (int u = 0; u < 8; ++u) { a[u] = lo[u]; a[u + 8] = hi[u]; }
    return a;
}

// A 16x32 bf16 fragment from an LDS bf16 row (forces ds_load_b128).
__device__ __forceinline__ v16bf load_a_l(const lbf16* row, int k0, int hl) {
    const lbf16* p = row + k0 + hl * 8;
    bf16x8 lo = *(const l_bf16x8*)p;
    bf16x8 hi = *(const l_bf16x8*)(p + 16);
    v16bf a;
#pragma unroll
    for (int u = 0; u < 8; ++u) { a[u] = lo[u]; a[u + 8] = hi[u]; }
    return a;
}

// B 32x16 bf16 fragment from the staged transposed LDS panel bsm[n][k] (stride BSTRIDE).
__device__ __forceinline__ v16bf load_b_frag(const lbf16* bsm, int j, int hl, int l16) {
    const lbf16* p = bsm + (j * 16 + l16) * BSTRIDE + hl * 8;
    bf16x8 lo = *(const l_bf16x8*)p;
    bf16x8 hi = *(const l_bf16x8*)(p + 16);
    v16bf b;
#pragma unroll
    for (int u = 0; u < 8; ++u) { b[u] = lo[u]; b[u + 8] = hi[u]; }
    return b;
}

// One K-step: load all 8 B fragments first, then 8 WMMAs (stages the ds waits).
__device__ __forceinline__ void compute_kstep(const lbf16* bsm, v16bf a, v8f acc[8],
                                              int hl, int l16) {
    v16bf bfr[8];
#pragma unroll
    for (int j = 0; j < 8; ++j) bfr[j] = load_b_frag(bsm, j, hl, l16);
#pragma unroll
    for (int j = 0; j < 8; ++j)
        acc[j] = __builtin_amdgcn_wmma_f32_16x16x32_bf16(
            false, a, false, bfr[j], (short)0, acc[j], false, false);
}

// Stage a 128(N) x 32(K) bf16 panel of Wt[e][n][k] into LDS (padded rows).
#if HAVE_ASYNC_LDS
__device__ __forceinline__ void stage_panel_async(const __bf16* wte, int k0, int nb,
                                                  lbf16* buf, int tid) {
    int n  = tid >> 1;
    int kh = (tid & 1) * 16;
    const __bf16* src = wte + (size_t)(nb + n) * DIM + k0 + kh;
    lbf16* dst = buf + n * BSTRIDE + kh;
    async_copy_b128(src, dst);
    async_copy_b128(src + 8, dst + 8);
}
#else
__device__ __forceinline__ void stage_panel_sync(const __bf16* wte, int k0, int nb,
                                                 lbf16* buf, int tid) {
    int n  = tid >> 1;
    int kh = (tid & 1) * 16;
    const __bf16* src = wte + (size_t)(nb + n) * DIM + k0 + kh;
    __builtin_prefetch(src + 32, 0, 1);   // next K panel
    bf16x8 v0 = *(const bf16x8*)src;
    bf16x8 v1 = *(const bf16x8*)(src + 8);
    lbf16* dst = buf + n * BSTRIDE + kh;
    *(l_bf16x8*)dst = v0;
    *(l_bf16x8*)(dst + 8) = v1;
}
#endif

// Full K loop (1024) for one 128-wide N chunk; double-buffered (static buffers) when async.
template <bool ALDS>
__device__ __forceinline__ void gemm_kloop(const __bf16* ag, const lbf16* al,
                                           const __bf16* wte, int nb,
                                           lbf16* buf0, lbf16* buf1, v8f acc[8],
                                           int tid, int hl, int l16) {
#if HAVE_ASYNC_LDS
    stage_panel_async(wte, 0, nb, buf0, tid);
#pragma unroll 1
    for (int kk = 0; kk < 16; ++kk) {
        int ks = kk * 2;
        __syncthreads();                               // buf1 readers of prev pair done
        stage_panel_async(wte, (ks + 1) * 32, nb, buf1, tid);
        wait_async_le2();                              // panel ks arrived (in-order)
        __syncthreads();
        {
            v16bf a = ALDS ? load_a_l(al, ks * 32, hl) : load_a_g(ag, ks * 32, hl);
            compute_kstep(buf0, a, acc, hl, l16);
        }
        __syncthreads();                               // buf0 readers done
        if (ks + 2 < 32) {
            stage_panel_async(wte, (ks + 2) * 32, nb, buf0, tid);
            wait_async_le2();                          // panel ks+1 arrived
        } else {
            wait_async_le0();
        }
        __syncthreads();
        {
            v16bf a = ALDS ? load_a_l(al, (ks + 1) * 32, hl) : load_a_g(ag, (ks + 1) * 32, hl);
            compute_kstep(buf1, a, acc, hl, l16);
        }
    }
#else
#pragma unroll 1
    for (int ks = 0; ks < 32; ++ks) {
        __syncthreads();
        stage_panel_sync(wte, ks * 32, nb, buf0, tid);
        __syncthreads();
        v16bf a = ALDS ? load_a_l(al, ks * 32, hl) : load_a_g(ag, ks * 32, hl);
        compute_kstep(buf0, a, acc, hl, l16);
    }
#endif
}

// ---------------- kernel 0: zero the expert counters ----------------
__global__ void moe_zero_counts(int* counts) {
    if (threadIdx.x < NE) counts[threadIdx.x] = 0;
}

// ---------------- kernel 1a: x f32 -> bf16 ----------------
__global__ __launch_bounds__(256) void moe_cvt_x(const float* __restrict__ x,
                                                 __bf16* __restrict__ xb) {
    size_t i = ((size_t)blockIdx.x * 256 + threadIdx.x) * 8;
    float4 a = *(const float4*)(x + i);
    float4 b = *(const float4*)(x + i + 4);
    bf16x8 o;
    o[0] = (__bf16)a.x; o[1] = (__bf16)a.y; o[2] = (__bf16)a.z; o[3] = (__bf16)a.w;
    o[4] = (__bf16)b.x; o[5] = (__bf16)b.y; o[6] = (__bf16)b.z; o[7] = (__bf16)b.w;
    *(bf16x8*)(xb + i) = o;
}

// ---------------- kernel 1b: W f32 [e][k][n] -> bf16 transposed Wt [e][n][k] ----------
__global__ __launch_bounds__(256) void moe_cvt_w(const float* __restrict__ w1,
                                                 const float* __restrict__ w2,
                                                 __bf16* __restrict__ wt1,
                                                 __bf16* __restrict__ wt2) {
    __shared__ float tile[64][65];
    const float* W = blockIdx.y ? w2 : w1;
    __bf16* WT = blockIdx.y ? wt2 : wt1;
    int b  = blockIdx.x;            // e*256 + kt*16 + nt
    int e  = b >> 8;
    int kt = (b >> 4) & 15;
    int nt = b & 15;
    const float* We = W + (size_t)e * DIM * DIM + (size_t)kt * 64 * DIM + nt * 64;
    __bf16* WTe = WT + (size_t)e * DIM * DIM + (size_t)nt * 64 * DIM + kt * 64;
    int r  = threadIdx.x >> 4;        // 0..15
    int c4 = (threadIdx.x & 15) * 4;  // 0..60
#pragma unroll
    for (int rr = 0; rr < 4; ++rr) {
        int row = rr * 16 + r;
        float4 v = *(const float4*)(We + (size_t)row * DIM + c4);
        tile[row][c4] = v.x; tile[row][c4 + 1] = v.y;
        tile[row][c4 + 2] = v.z; tile[row][c4 + 3] = v.w;
    }
    __syncthreads();
#pragma unroll
    for (int rr = 0; rr < 4; ++rr) {
        int n = rr * 16 + r;
        bf16x4 o;
        o[0] = (__bf16)tile[c4][n];     o[1] = (__bf16)tile[c4 + 1][n];
        o[2] = (__bf16)tile[c4 + 2][n]; o[3] = (__bf16)tile[c4 + 3][n];
        *(bf16x4*)(WTe + (size_t)n * DIM + c4) = o;
    }
}

// ---------------- kernel 2: gating (one wave per token) ----------------
__global__ __launch_bounds__(256) void moe_gate(const float* __restrict__ x,
                                                const float* __restrict__ gw,
                                                const float* __restrict__ gb,
                                                int* counts, int* btok, float* bwt, int* bslot) {
    int wave = threadIdx.x >> 5;
    int lane = threadIdx.x & 31;
    int t = blockIdx.x * 8 + wave;
    if (t >= NTOK) return;

    const float* xr = x + (size_t)t * DIM;
    float acc[NE];
#pragma unroll
    for (int e = 0; e < NE; ++e) acc[e] = 0.f;

    for (int d = lane; d < DIM; d += 32) {
        float xv = xr[d];
        const float4* g4 = (const float4*)(gw + (size_t)d * NE);
        float4 g0 = g4[0], g1 = g4[1];
        acc[0] += xv * g0.x; acc[1] += xv * g0.y; acc[2] += xv * g0.z; acc[3] += xv * g0.w;
        acc[4] += xv * g1.x; acc[5] += xv * g1.y; acc[6] += xv * g1.z; acc[7] += xv * g1.w;
    }
#pragma unroll
    for (int off = 16; off > 0; off >>= 1) {
#pragma unroll
        for (int e = 0; e < NE; ++e) acc[e] += __shfl_xor(acc[e], off, 32);
    }

    if (lane == 0) {
        float l[NE];
        float m = -INFINITY;
#pragma unroll
        for (int e = 0; e < NE; ++e) { l[e] = acc[e] + gb[e]; m = fmaxf(m, l[e]); }
        float s = 0.f;
#pragma unroll
        for (int e = 0; e < NE; ++e) { l[e] = expf(l[e] - m); s += l[e]; }
        float inv = 1.f / s;
#pragma unroll
        for (int e = 0; e < NE; ++e) l[e] *= inv;        // softmax probs

        int i0 = 0; float p0 = l[0];
#pragma unroll
        for (int e = 1; e < NE; ++e) if (l[e] > p0) { p0 = l[e]; i0 = e; }
        int i1 = -1; float p1 = -INFINITY;
#pragma unroll
        for (int e = 0; e < NE; ++e) if (e != i0 && l[e] > p1) { p1 = l[e]; i1 = e; }

        // softmax over the two selected probabilities (as in reference)
        float e1 = expf(p1 - p0);
        float wsum = 1.f / (1.f + e1);
        float w0 = wsum, w1 = e1 * wsum;

        int pos0 = atomicAdd(&counts[i0], 1);
        int idx0 = i0 * NTOK + pos0;
        btok[idx0] = t; bwt[idx0] = w0; bslot[idx0] = 0;

        int pos1 = atomicAdd(&counts[i1], 1);
        int idx1 = i1 * NTOK + pos1;
        btok[idx1] = t; bwt[idx1] = w1; bslot[idx1] = 1;
    }
}

// ---------------- kernel 3: fused per-expert MLP (WMMA bf16) ----------------
__global__ __launch_bounds__(256) void moe_expert(const __bf16* __restrict__ xb,
                                                  const __bf16* __restrict__ wt1,
                                                  const __bf16* __restrict__ wt2,
                                                  const float* __restrict__ b1,
                                                  const float* __restrict__ b2,
                                                  const int* __restrict__ counts,
                                                  const int* __restrict__ btok,
                                                  const float* __restrict__ bwt,
                                                  const int* __restrict__ bslot,
                                                  float* __restrict__ yp) {
    extern __shared__ __bf16 smem[];
    lbf16* hsm  = (lbf16*)smem;                    // [TILE_M][HSTRIDE] bf16 h tile
    lbf16* bsm0 = hsm + TILE_M * HSTRIDE;          // panel buffer 0
    lbf16* bsm1 = bsm0 + 128 * BSTRIDE;            // panel buffer 1

    int e    = blockIdx.x >> 6;
    int tile = blockIdx.x & 63;
    int cnt  = counts[e];
    int base = tile * TILE_M;
    if (base >= cnt) return;        // uniform exit, before any barrier

    int tid  = threadIdx.x;
    int wave = tid >> 5;
    int lane = tid & 31;
    int hl   = lane >> 4;
    int l16  = lane & 15;
    int m0   = wave * 16;

    const __bf16* wt1e = wt1 + (size_t)e * DIM * DIM;
    const __bf16* wt2e = wt2 + (size_t)e * DIM * DIM;
    const float*  b1e  = b1 + (size_t)e * DIM;
    const float*  b2e  = b2 + (size_t)e * DIM;

    // A-row token (M = m0 + l16)
    int aidx = base + m0 + l16;
    int tka  = (aidx < cnt) ? btok[e * NTOK + aidx] : 0;
    const __bf16* xrow = xb + (size_t)tka * DIM;

    // C/D row metadata (M = m0 + hl*8 + i)
    int tokm[8]; float wm[8]; int slm[8]; bool vm[8];
#pragma unroll
    for (int i = 0; i < 8; ++i) {
        int g = base + m0 + hl * 8 + i;
        vm[i]   = g < cnt;
        int gi  = e * NTOK + (vm[i] ? g : 0);
        tokm[i] = vm[i] ? btok[gi] : 0;
        wm[i]   = vm[i] ? bwt[gi] : 0.f;
        slm[i]  = vm[i] ? bslot[gi] : 0;
    }

    // -------- phase 1: h = relu(x @ W1 + b1) -> LDS (bf16) --------
    for (int nc = 0; nc < NCH; ++nc) {
        int nb = nc * 128;
        v8f acc[8];
#pragma unroll
        for (int j = 0; j < 8; ++j) {
            float bias = b1e[nb + j * 16 + l16];
#pragma unroll
            for (int i = 0; i < 8; ++i) acc[j][i] = bias;
        }
        gemm_kloop<false>(xrow, hsm, wt1e, nb, bsm0, bsm1, acc, tid, hl, l16);
#pragma unroll
        for (int j = 0; j < 8; ++j) {
            int n = nb + j * 16 + l16;
#pragma unroll
            for (int i = 0; i < 8; ++i) {
                float v = acc[j][i];
                v = v > 0.f ? v : 0.f;             // relu
                hsm[(m0 + hl * 8 + i) * HSTRIDE + n] = (__bf16)v;
            }
        }
    }

    // -------- phase 2: y = (h @ W2 + b2) * gate_weight -> yp --------
    const lbf16* harow = hsm + (m0 + l16) * HSTRIDE;
    for (int nc = 0; nc < NCH; ++nc) {
        int nb = nc * 128;
        v8f acc[8];
#pragma unroll
        for (int j = 0; j < 8; ++j) {
            float bias = b2e[nb + j * 16 + l16];
#pragma unroll
            for (int i = 0; i < 8; ++i) acc[j][i] = bias;
        }
        gemm_kloop<true>(xrow, harow, wt2e, nb, bsm0, bsm1, acc, tid, hl, l16);
#pragma unroll
        for (int j = 0; j < 8; ++j) {
            int n = nb + j * 16 + l16;
#pragma unroll
            for (int i = 0; i < 8; ++i) {
                if (vm[i]) {
                    size_t o = ((size_t)slm[i] * NTOK + tokm[i]) * DIM + n;
                    yp[o] = acc[j][i] * wm[i];
                }
            }
        }
    }
}

// ---------------- kernel 4: combine the two expert slots ----------------
__global__ void moe_combine(const float* __restrict__ yp, float* __restrict__ out) {
    size_t i = (size_t)blockIdx.x * blockDim.x + threadIdx.x;   // float4 index
    const float4* a = (const float4*)yp;
    float4 u = a[i];
    float4 v = a[i + (size_t)NTOK * DIM / 4];
    float4 r; r.x = u.x + v.x; r.y = u.y + v.y; r.z = u.z + v.z; r.w = u.w + v.w;
    ((float4*)out)[i] = r;
}

// ---------------- launch ----------------
extern "C" void kernel_launch(void* const* d_in, const int* in_sizes, int n_in,
                              void* d_out, int out_size, void* d_ws, size_t ws_size,
                              hipStream_t stream) {
    const float* x  = (const float*)d_in[0];
    const float* gw = (const float*)d_in[1];
    const float* gb = (const float*)d_in[2];
    const float* w1 = (const float*)d_in[3];
    const float* b1 = (const float*)d_in[4];
    const float* w2 = (const float*)d_in[5];
    const float* b2 = (const float*)d_in[6];
    float* out = (float*)d_out;

    // workspace layout
    char* ws = (char*)d_ws;
    size_t o = 0;
    int*    counts = (int*)(ws + o);   o += 256;
    int*    btok   = (int*)(ws + o);   o += 4ull * NE * NTOK;
    float*  bwt    = (float*)(ws + o); o += 4ull * NE * NTOK;
    int*    bslot  = (int*)(ws + o);   o += 4ull * NE * NTOK;
    float*  yp     = (float*)(ws + o); o += 4ull * KTOP * NTOK * DIM;
    __bf16* xbp    = (__bf16*)(ws + o); o += 2ull * NTOK * DIM;
    __bf16* wt1    = (__bf16*)(ws + o); o += 2ull * NE * DIM * DIM;
    __bf16* wt2    = (__bf16*)(ws + o); o += 2ull * NE * DIM * DIM;

    moe_zero_counts<<<1, 32, 0, stream>>>(counts);

    moe_cvt_x<<<(NTOK * DIM) / (256 * 8), 256, 0, stream>>>(x, xbp);
    moe_cvt_w<<<dim3(NE * 256, 2), 256, 0, stream>>>(w1, w2, wt1, wt2);

    moe_gate<<<NTOK / 8, 256, 0, stream>>>(x, gw, gb, counts, btok, bwt, bslot);

    size_t smem = ((size_t)TILE_M * HSTRIDE + 2ull * 128 * BSTRIDE) * sizeof(__bf16);
    moe_expert<<<NE * NTILES, 256, smem, stream>>>(xbp, wt1, wt2, b1, b2,
                                                   counts, btok, bwt, bslot, yp);

    moe_combine<<<(NTOK * (DIM / 4)) / 256, 256, 0, stream>>>(yp, out);
}
